// GRU_Model_10617159155731
// MI455X (gfx1250) — matrix operand order repaired and checked by reading, not compiled, not run on hardware
//
#include <hip/hip_runtime.h>
#include <hip/hip_bf16.h>

// ---------------------------------------------------------------------------
// Fused GRU for MI455X (gfx1250, wave32, WMMA).
//
// Each wave owns 16 batch rows for the full T=128 recurrence:
//   per step:  xp = x_t @ W + b0   (16x64 * 64x192, bf16 WMMA, fp32 acc)
//              hp = h   @ U + b1   (16x64 * 64x192)
//              z=sig(xz+hz) r=sig(xr+hr) hc=relu(xh+r*hh)
//              h = z*h + (1-z)*hc
//   epilogue:  out = h @ Wd + bd   (shuffle reduction)
//
// Gate blocks padded 50 -> 64 (N=192, 12 tiles) so z/r/h of a unit share a
// lane. U B-fragments are pinned in VGPRs (reused 128x, on the recurrence
// critical path); W B-fragments are streamed from LDS (overlap with h@U
// WMMAs). h transposed D->A layout through per-wave LDS scratch (in-order DS
// within a wave; wave_barrier for compiler ordering).
// ---------------------------------------------------------------------------

typedef __attribute__((ext_vector_type(16))) __bf16 bf16x16;
typedef __attribute__((ext_vector_type(8)))  __bf16 bf16x8;
typedef __attribute__((ext_vector_type(8)))  float  f32x8;
typedef __attribute__((ext_vector_type(4)))  float  f32x4;

#define GRU_UNITS 50
#define GRU_T     128
#define GRU_F     64
#define GRU_NT    12        // N tiles of 16 (3 gates x 64 pad)
#define GRU_K3U   (3 * GRU_UNITS)
#define GRU_B     4096
#define ROWS_PER_WAVE 16
#define WAVES_PER_WG  2
#define WG_THREADS    (WAVES_PER_WG * 32)
#define ROWS_PER_WG   (WAVES_PER_WG * ROWS_PER_WAVE)
#define FRAG_ELEMS    (2 * GRU_NT * 32 * 16)   // chunks * tiles * lanes * elems

__device__ __forceinline__ float sigmoid_fast(float v) {
    return 1.0f / (1.0f + __expf(-v));
}

#define WMMA_BF16(A, B, C) \
    __builtin_amdgcn_wmma_f32_16x16x32_bf16(false, (A), false, (B), (short)0, (C), false, false)

__global__ __launch_bounds__(WG_THREADS, 1)
void gru_fused_kernel(const float* __restrict__ x,
                      const float* __restrict__ W,
                      const float* __restrict__ U,
                      const float* __restrict__ b,
                      const float* __restrict__ Wd,
                      const float* __restrict__ bd,
                      float* __restrict__ out) {
    // LDS: W in exact B-fragment order (bf16) + per-wave h transpose scratch.
    __shared__ __align__(32) __bf16 Wfrag[FRAG_ELEMS];           // 24 KB
    __shared__ __align__(16) __bf16 hscr[WAVES_PER_WG][16 * 64]; // 2 KB/wave

    const int tid  = threadIdx.x;
    const int lane = tid & 31;
    const int wave = tid >> 5;

    // ---- one-time: pack W into B-fragment layout (zero-padded) ----
    // B fragment (16x16x32 bf16): lane L holds column n = 16*t + (L&15),
// K values k = 32*c + (L&16)/2 + (e&7) + 16*(e>>3), e = 0..15, stored lane-contiguous.
    for (int idx = tid; idx < FRAG_ELEMS; idx += WG_THREADS) {
        const int e  = idx & 15;
        const int ln = (idx >> 4) & 31;
        const int t  = (idx >> 9) % GRU_NT;
        const int c  = idx / (GRU_NT * 512);
        const int k  = 32 * c + ((ln & 16) >> 1) + (e & 7) + ((e & 8) << 1);      // 0..63
        const int n  = 16 * t + (ln & 15);          // 0..191
        const int g  = n >> 6;                      // gate 0..2
        const int u  = n & 63;                      // unit within gate (padded)
        float wv = 0.0f;
        if (u < GRU_UNITS) wv = W[k * GRU_K3U + g * GRU_UNITS + u];
        Wfrag[idx] = (__bf16)wv;
    }
    __syncthreads();

    const int mrow  = lane & 15;             // A-row / D-column within tile
    const int half8 = (lane & 16) >> 1;      // 0 or 8: D-row offset, A K-subgroup
    const int rowBase = blockIdx.x * ROWS_PER_WG + wave * ROWS_PER_WAVE;

    // ---- U B-fragments pinned in VGPRs (reused every step) ----
    bf16x16 uf0[GRU_NT], uf1[GRU_NT];
#pragma unroll
    for (int t = 0; t < GRU_NT; ++t) {
        const int n = 16 * t + mrow;
        const int g = n >> 6, u = n & 63;
        const bool nv = (u < GRU_UNITS);
        const int col = g * GRU_UNITS + u;
#pragma unroll
        for (int e = 0; e < 16; ++e) {
            const int k0 = ((lane & 16) >> 1) + (e & 7) + ((e & 8) << 1);          // chunk 0: K 0..31
            const int k1 = 32 + ((lane & 16) >> 1) + (e & 7) + ((e & 8) << 1);     // chunk 1: K 32..63
            uf0[t][e] = (__bf16)((nv && k0 < GRU_UNITS) ? U[k0 * GRU_K3U + col] : 0.0f);
            uf1[t][e] = (__bf16)((nv && k1 < GRU_UNITS) ? U[k1 * GRU_K3U + col] : 0.0f);
        }
    }

    // ---- per-lane bias registers (packed-column layout) ----
    float b0r[GRU_NT], b1r[GRU_NT];
#pragma unroll
    for (int t = 0; t < GRU_NT; ++t) {
        const int n = 16 * t + mrow;
        const int g = n >> 6, u = n & 63;
        const bool v = (u < GRU_UNITS);
        b0r[t] = v ? b[g * GRU_UNITS + u] : 0.0f;
        b1r[t] = v ? b[GRU_K3U + g * GRU_UNITS + u] : 0.0f;
    }
    float wdr[4];
#pragma unroll
    for (int tl = 0; tl < 4; ++tl) {
        const int c = 16 * tl + mrow;
        wdr[tl] = (c < GRU_UNITS) ? Wd[c] : 0.0f;
    }
    const float bd0 = bd[0];

    // ---- recurrent state ----
    f32x8 htile[4];
#pragma unroll
    for (int tl = 0; tl < 4; ++tl)
#pragma unroll
        for (int j = 0; j < 8; ++j) htile[tl][j] = 0.0f;
    bf16x16 ah0 = {}, ah1 = {};              // h A-fragments (h0 = 0)
    __bf16* myscr = &hscr[wave][0];

    const float* xlane = x + (size_t)(rowBase + mrow) * (GRU_T * GRU_F);

#pragma unroll 1
    for (int t = 0; t < GRU_T; ++t) {
        // ---- x_t A-fragments: lane reads its row, 4x 8-float runs ----
        const float* xp = xlane + t * GRU_F;
        bf16x16 ax0, ax1;
        {
            f32x4 q0 = *(const f32x4*)(xp + half8);
            f32x4 q1 = *(const f32x4*)(xp + half8 + 4);
            f32x4 q2 = *(const f32x4*)(xp + half8 + 16);
            f32x4 q3 = *(const f32x4*)(xp + half8 + 20);
            f32x4 q4 = *(const f32x4*)(xp + half8 + 32);
            f32x4 q5 = *(const f32x4*)(xp + half8 + 36);
            f32x4 q6 = *(const f32x4*)(xp + half8 + 48);
            f32x4 q7 = *(const f32x4*)(xp + half8 + 52);
#pragma unroll
            for (int e = 0; e < 4; ++e) {
                ax0[e]      = (__bf16)q0[e];  ax0[4 + e]  = (__bf16)q1[e];
                ax0[8 + e]  = (__bf16)q2[e];  ax0[12 + e] = (__bf16)q3[e];
                ax1[e]      = (__bf16)q4[e];  ax1[4 + e]  = (__bf16)q5[e];
                ax1[8 + e]  = (__bf16)q6[e];  ax1[12 + e] = (__bf16)q7[e];
            }
        }

        // ---- accumulators seeded with biases ----
        // tiles 0..7 (z,r gates): x- and h-GEMMs share one accumulator.
        f32x8 acc[GRU_NT], acch[4];
#pragma unroll
        for (int n = 0; n < GRU_NT; ++n) {
            const float s = b0r[n] + ((n < 8) ? b1r[n] : 0.0f);
#pragma unroll
            for (int j = 0; j < 8; ++j) acc[n][j] = s;
        }
#pragma unroll
        for (int tl = 0; tl < 4; ++tl)
#pragma unroll
            for (int j = 0; j < 8; ++j) acch[tl][j] = b1r[tl + 8];

        // ---- hp = h @ U : 24 WMMAs, B operands register-resident ----
#pragma unroll
        for (int n = 0; n < 8; ++n) {
            acc[n] = WMMA_BF16(ah0, uf0[n], acc[n]);
            acc[n] = WMMA_BF16(ah1, uf1[n], acc[n]);
        }
#pragma unroll
        for (int tl = 0; tl < 4; ++tl) {
            acch[tl] = WMMA_BF16(ah0, uf0[tl + 8], acch[tl]);
            acch[tl] = WMMA_BF16(ah1, uf1[tl + 8], acch[tl]);
        }

        // ---- xp = x @ W : 24 WMMAs, B operands streamed from LDS ----
#pragma unroll
        for (int n = 0; n < GRU_NT; ++n) {
            bf16x16 wb0 = *(const bf16x16*)&Wfrag[((0 * GRU_NT + n) * 32 + lane) * 16];
            acc[n] = WMMA_BF16(ax0, wb0, acc[n]);
            bf16x16 wb1 = *(const bf16x16*)&Wfrag[((1 * GRU_NT + n) * 32 + lane) * 16];
            acc[n] = WMMA_BF16(ax1, wb1, acc[n]);
        }

        // ---- gates (per-lane: z,r,h of a unit share lane across tiles) ----
        __builtin_amdgcn_wave_barrier();
#pragma unroll
        for (int tl = 0; tl < 4; ++tl) {
#pragma unroll
            for (int j = 0; j < 8; ++j) {
                const float z  = sigmoid_fast(acc[tl][j]);
                const float r  = sigmoid_fast(acc[tl + 4][j]);
                const float hc = fmaxf(fmaf(r, acch[tl][j], acc[tl + 8][j]), 0.0f);
                const float hn = fmaf(z, htile[tl][j] - hc, hc); // z*h+(1-z)*hc
                htile[tl][j] = hn;
                // D-layout element (row j+half8, col 16*tl+mrow) -> row-major scratch
                myscr[(j + half8) * 64 + 16 * tl + mrow] = (__bf16)hn;
            }
        }
        __builtin_amdgcn_wave_barrier();

        // ---- reload h as A-fragments (transpose via per-wave LDS) ----
        {
            bf16x8 p0 = *(const bf16x8*)&myscr[mrow * 64 + 0  + half8];
            bf16x8 p1 = *(const bf16x8*)&myscr[mrow * 64 + 16 + half8];
            bf16x8 p2 = *(const bf16x8*)&myscr[mrow * 64 + 32 + half8];
            bf16x8 p3 = *(const bf16x8*)&myscr[mrow * 64 + 48 + half8];
#pragma unroll
            for (int e = 0; e < 8; ++e) {
                ah0[e] = p0[e]; ah0[8 + e] = p1[e];
                ah1[e] = p2[e]; ah1[8 + e] = p3[e];
            }
        }
        __builtin_amdgcn_wave_barrier();
    }

    // ---- epilogue: out = hT @ Wd + bd (cross-lane dot over 16 columns) ----
    float part[8];
#pragma unroll
    for (int j = 0; j < 8; ++j) {
        part[j] = htile[0][j] * wdr[0] + htile[1][j] * wdr[1]
                + htile[2][j] * wdr[2] + htile[3][j] * wdr[3];
    }
#pragma unroll
    for (int m = 1; m < 16; m <<= 1) {
#pragma unroll
        for (int j = 0; j < 8; ++j)
            part[j] += __shfl_xor(part[j], m, 32);
    }
    if (mrow == 0) {
#pragma unroll
        for (int j = 0; j < 8; ++j)
            out[rowBase + half8 + j] = part[j] + bd0;
    }
}

extern "C" void kernel_launch(void* const* d_in, const int* in_sizes, int n_in,
                              void* d_out, int out_size, void* d_ws, size_t ws_size,
                              hipStream_t stream) {
    const float* x  = (const float*)d_in[0];
    const float* W  = (const float*)d_in[1];
    const float* U  = (const float*)d_in[2];
    const float* b  = (const float*)d_in[3];
    const float* Wd = (const float*)d_in[4];
    const float* bd = (const float*)d_in[5];
    float* out = (float*)d_out;

    dim3 grid(GRU_B / ROWS_PER_WG);   // 128 workgroups
    dim3 block(WG_THREADS);           // 2 waves, 16 batch rows each
    gru_fused_kernel<<<grid, block, 0, stream>>>(x, W, U, b, Wd, bd, out);
}
